// MLASegmentCrossAttention_14456859919082
// MI455X (gfx1250) — matrix operand order repaired and checked by reading, not compiled, not run on hardware
//
#include <hip/hip_runtime.h>

// ---------------------------------------------------------------------------
// MLA segment cross-attention for MI455X (gfx1250).
// All dense GEMMs run through v_wmma_f32_16x16x32_bf16 (wave32 WMMA) with
// double-buffered LDS pipelining; attention itself is 9-sparse (LOOKBACK=8)
// so it is a cheap VALU epilogue.
// ---------------------------------------------------------------------------

typedef __bf16 bf16_t;
typedef __attribute__((ext_vector_type(8)))  __bf16 v8bf;
typedef __attribute__((ext_vector_type(16))) __bf16 v16bf;
typedef __attribute__((ext_vector_type(8)))  float  v8f;

#define B_    2
#define LQ_   4096
#define LK_   1024
#define QDIM_ 2048
#define KVDIM_ 2048
#define H_    16
#define HD_   128
#define KVC_  512
#define QC_   768
#define R_    64
#define LOOKBACK_ 8
#define NEG_  (-1e30f)

__device__ __forceinline__ bf16_t f2bf(float f) {
  unsigned u = __builtin_bit_cast(unsigned, f);
  unsigned r = u + 0x7FFFu + ((u >> 16) & 1u);   // round-to-nearest-even
  unsigned short h = (unsigned short)(r >> 16);
  return __builtin_bit_cast(bf16_t, h);
}
__device__ __forceinline__ float bf2f(bf16_t b) {
  unsigned short h = __builtin_bit_cast(unsigned short, b);
  unsigned u = ((unsigned)h) << 16;
  return __builtin_bit_cast(float, u);
}

// ------------------------------- casts -------------------------------------
__global__ __launch_bounds__(256) void cast_f32_bf16(const float* __restrict__ in,
                                                     bf16_t* __restrict__ out, long n) {
  long i = (long)blockIdx.x * blockDim.x + threadIdx.x;
  if (i < n) out[i] = f2bf(in[i]);
}

// W[K][N] fp32 -> WT[N][K] bf16 (so GEMM B-operand is K-contiguous per column)
__global__ __launch_bounds__(256) void transpose_cast(const float* __restrict__ W,
                                                      bf16_t* __restrict__ WT,
                                                      int K, int N) {
  long i = (long)blockIdx.x * blockDim.x + threadIdx.x;
  long tot = (long)K * N;
  if (i < tot) {
    int k = (int)(i / N);
    int n = (int)(i % N);
    WT[(size_t)n * K + k] = f2bf(W[i]);
  }
}

// ------------------------------- WMMA GEMM ---------------------------------
// C[M,N] = A[M,K](bf16,row-major) x Bt[N,K](bf16,N-major i.e. B^T row-major)
// 256 threads = 8 waves; block tile 128x128x32; wave tile 32x64 (2x4 WMMAs).
// Double-buffered LDS: global loads for tile t+1 overlap WMMA on tile t,
// one s_barrier per k-step. Staging loads are branch-free (row clamping:
// OOB rows only feed never-stored output rows/cols).
#define TILE_M 128
#define TILE_N 128
#define TILE_K 32
#define LDSS   40   // bf16 elems/row: 32 + 8 pad = 80B (16B aligned, bank-friendly)
#define ABUF   (TILE_M * LDSS)
#define BBUF   (TILE_N * LDSS)

template <bool OUT_BF16>
__global__ __launch_bounds__(256)
void gemm_wmma_bf16(const bf16_t* __restrict__ A, const bf16_t* __restrict__ Bt,
                    void* __restrict__ Cout, int M, int N, int K) {
  __shared__ bf16_t sA[2 * ABUF];
  __shared__ bf16_t sB[2 * BBUF];

  const int tid  = threadIdx.x;
  const int lane = tid & 31;
  const int wave = tid >> 5;
  const int half = lane >> 4;   // 0|1
  const int l16  = lane & 15;

  const int bm = blockIdx.y * TILE_M;
  const int bn = blockIdx.x * TILE_N;
  const int wr = (wave >> 1) * 32;   // wave row offset (4 wave-rows)
  const int wc = (wave & 1) * 64;    // wave col offset (2 wave-cols)

  // ---- loop-invariant staging addresses (row-clamped, branch-free) ----
  const int r0 = tid >> 2;            // 0..63
  const int r1 = r0 + 64;             // 64..127
  const int c  = (tid & 3) << 3;      // 0,8,16,24
  const int ma0 = min(bm + r0, M - 1);
  const int ma1 = min(bm + r1, M - 1);
  const int nb0 = min(bn + r0, N - 1);
  const int nb1 = min(bn + r1, N - 1);
  const bf16_t* pa0 = A  + (size_t)ma0 * K + c;
  const bf16_t* pa1 = A  + (size_t)ma1 * K + c;
  const bf16_t* pb0 = Bt + (size_t)nb0 * K + c;
  const bf16_t* pb1 = Bt + (size_t)nb1 * K + c;
  const int sOffA0 = r0 * LDSS + c;
  const int sOffA1 = r1 * LDSS + c;

  v8f acc[2][4];
#pragma unroll
  for (int i = 0; i < 2; ++i)
#pragma unroll
    for (int j = 0; j < 4; ++j)
#pragma unroll
      for (int e = 0; e < 8; ++e) acc[i][j][e] = 0.f;

  // ---- prologue: stage tile 0 into buffer 0 ----
  v8bf ra0 = *(const v8bf*)pa0;
  v8bf ra1 = *(const v8bf*)pa1;
  v8bf rb0 = *(const v8bf*)pb0;
  v8bf rb1 = *(const v8bf*)pb1;
  *(v8bf*)(sA + sOffA0) = ra0;
  *(v8bf*)(sA + sOffA1) = ra1;
  *(v8bf*)(sB + sOffA0) = rb0;
  *(v8bf*)(sB + sOffA1) = rb1;
  __syncthreads();

  const int nT = K / TILE_K;   // K is always a multiple of 32 here
  int buf = 0;
  for (int t = 0; t < nT; ++t) {
    const bool more = (t + 1) < nT;
    if (more) {   // uniform branch; issue next-tile global loads early
      const int ko = (t + 1) * TILE_K;
      ra0 = *(const v8bf*)(pa0 + ko);
      ra1 = *(const v8bf*)(pa1 + ko);
      rb0 = *(const v8bf*)(pb0 + ko);
      rb1 = *(const v8bf*)(pb1 + ko);
    }

    // ---- fragments per ISA 7.12.2 layouts ----
    const bf16_t* bA = sA + buf * ABUF;
    const bf16_t* bB = sB + buf * BBUF;
    v16bf af[2], bfr[4];
#pragma unroll
    for (int i = 0; i < 2; ++i) {
      const bf16_t* p = bA + (wr + i * 16 + l16) * LDSS;
      v8bf lo = *(const v8bf*)(p + half * 8);        // K 0-7 | 8-15
      v8bf hi = *(const v8bf*)(p + 16 + half * 8);   // K 16-23 | 24-31
      af[i] = __builtin_shufflevector(lo, hi, 0,1,2,3,4,5,6,7,8,9,10,11,12,13,14,15);
    }
#pragma unroll
    for (int j = 0; j < 4; ++j) {
      const bf16_t* p = bB + (wc + j * 16 + l16) * LDSS;
      v8bf lo = *(const v8bf*)(p + half * 8);        // K 0-7 | 8-15
      v8bf hi = *(const v8bf*)(p + 16 + half * 8);   // K 16-23 | 24-31
      bfr[j] = __builtin_shufflevector(lo, hi, 0,1,2,3,4,5,6,7,8,9,10,11,12,13,14,15);
    }
#pragma unroll
    for (int i = 0; i < 2; ++i)
#pragma unroll
      for (int j = 0; j < 4; ++j)
        acc[i][j] = __builtin_amdgcn_wmma_f32_16x16x32_bf16(
            false, af[i], false, bfr[j], (short)0, acc[i][j], false, false);

    if (more) {   // stage tile t+1 into the other buffer; one barrier per step
      const int nb = buf ^ 1;
      *(v8bf*)(sA + nb * ABUF + sOffA0) = ra0;
      *(v8bf*)(sA + nb * ABUF + sOffA1) = ra1;
      *(v8bf*)(sB + nb * BBUF + sOffA0) = rb0;
      *(v8bf*)(sB + nb * BBUF + sOffA1) = rb1;
      __syncthreads();
      buf = nb;
    }
  }

  // ---- store: C layout lanes0-15 M=r, lanes16-31 M=8+r, N=l16 ----
#pragma unroll
  for (int i = 0; i < 2; ++i) {
#pragma unroll
    for (int j = 0; j < 4; ++j) {
      int tm = bm + wr + i * 16 + half * 8;
      int tn = bn + wc + j * 16 + l16;
      if (tn < N) {
#pragma unroll
        for (int r = 0; r < 8; ++r) {
          int m = tm + r;
          if (m < M) {
            if (OUT_BF16)
              ((bf16_t*)Cout)[(size_t)m * N + tn] = f2bf(acc[i][j][r]);
            else
              ((float*)Cout)[(size_t)m * N + tn] = acc[i][j][r];
          }
        }
      }
    }
  }
}

// ------------------------------- RoPE (in-place, bf16) ---------------------
// rows of length 64, half=32; pos = (row / pos_div) % pos_mod
__global__ __launch_bounds__(256) void rope_kernel(bf16_t* __restrict__ x,
                                                   int rows, int pos_div, int pos_mod) {
  int i = blockIdx.x * blockDim.x + threadIdx.x;
  int row = i >> 5;
  int j = i & 31;
  if (row >= rows) return;
  int pos = (row / pos_div) % pos_mod;
  float inv = __powf(10000.f, -(float)j * (1.f / 32.f));
  float ang = (float)pos * inv;
  float s, cc;
  __sincosf(ang, &s, &cc);   // s = sin(ang), cc = cos(ang)
  bf16_t* p = x + (size_t)row * 64;
  float x1 = bf2f(p[j]);
  float x2 = bf2f(p[j + 32]);
  p[j]      = f2bf(x1 * cc - x2 * s);
  p[j + 32] = f2bf(x1 * s + x2 * cc);
}

// ------------------------------- sparse attention --------------------------
// one wave per (b, q, h); <=9 keys; fp32 math, bf16 in/out
__global__ __launch_bounds__(256)
void attn_sparse(const bf16_t* __restrict__ qn, const bf16_t* __restrict__ qr,
                 const bf16_t* __restrict__ kn, const bf16_t* __restrict__ kr,
                 const bf16_t* __restrict__ vv, const int* __restrict__ seg,
                 bf16_t* __restrict__ out) {
  int g = blockIdx.x * blockDim.x + threadIdx.x;
  int w = g >> 5;
  int lane = g & 31;
  if (w >= B_ * LQ_ * H_) return;            // exact fit; uniform per wave
  int h  = w & (H_ - 1);
  int bq = w >> 4;                           // H_ == 16
  int b  = bq >> 12;                         // LQ_ == 4096

  int s = seg[bq];
  int klo = s - LOOKBACK_; if (klo < 0) klo = 0;
  int cnt = s - klo + 1;

  const bf16_t* qnp = qn + (size_t)bq * (H_ * HD_) + h * HD_ + lane * 4;
  const bf16_t* qrp = qr + (size_t)bq * (H_ * R_) + h * R_ + lane * 2;
  float qv0 = bf2f(qnp[0]), qv1 = bf2f(qnp[1]), qv2 = bf2f(qnp[2]), qv3 = bf2f(qnp[3]);
  float qr0 = bf2f(qrp[0]), qr1 = bf2f(qrp[1]);

  const float scale = 0.07216878364870323f;  // 1/sqrt(HD + R) = 1/sqrt(192)
  float sc[LOOKBACK_ + 1];
#pragma unroll
  for (int kk = 0; kk <= LOOKBACK_; ++kk) {
    float p = 0.f;
    if (kk < cnt) {
      int k = klo + kk;
      const bf16_t* knp = kn + (size_t)(b * LK_ + k) * (H_ * HD_) + h * HD_ + lane * 4;
      const bf16_t* krp = kr + (size_t)(b * LK_ + k) * R_ + lane * 2;
      p = qv0 * bf2f(knp[0]) + qv1 * bf2f(knp[1]) + qv2 * bf2f(knp[2]) + qv3 * bf2f(knp[3])
        + qr0 * bf2f(krp[0]) + qr1 * bf2f(krp[1]);
    }
#pragma unroll
    for (int off = 16; off >= 1; off >>= 1) p += __shfl_xor(p, off, 32);
    sc[kk] = (kk < cnt) ? p * scale : NEG_;
  }
  float m = sc[0];
#pragma unroll
  for (int kk = 1; kk <= LOOKBACK_; ++kk) m = fmaxf(m, sc[kk]);
  float pr[LOOKBACK_ + 1], sum = 0.f;
#pragma unroll
  for (int kk = 0; kk <= LOOKBACK_; ++kk) { pr[kk] = __expf(sc[kk] - m); sum += pr[kk]; }
  float inv = 1.f / sum;

  float o0 = 0.f, o1 = 0.f, o2 = 0.f, o3 = 0.f;
  for (int kk = 0; kk < cnt; ++kk) {
    int k = klo + kk;
    const bf16_t* vp = vv + (size_t)(b * LK_ + k) * (H_ * HD_) + h * HD_ + lane * 4;
    float pw = pr[kk] * inv;
    o0 += pw * bf2f(vp[0]); o1 += pw * bf2f(vp[1]);
    o2 += pw * bf2f(vp[2]); o3 += pw * bf2f(vp[3]);
  }
  bf16_t* op = out + (size_t)bq * (H_ * HD_) + h * HD_ + lane * 4;
  op[0] = f2bf(o0); op[1] = f2bf(o1); op[2] = f2bf(o2); op[3] = f2bf(o3);
}

// ------------------------------- launcher ----------------------------------
extern "C" void kernel_launch(void* const* d_in, const int* in_sizes, int n_in,
                              void* d_out, int out_size, void* d_ws, size_t ws_size,
                              hipStream_t stream) {
  (void)in_sizes; (void)n_in; (void)out_size; (void)ws_size;
  const float* q     = (const float*)d_in[0];
  const float* kv    = (const float*)d_in[1];
  const int*   seg   = (const int*)d_in[2];
  const float* W_kvc = (const float*)d_in[3];
  const float* W_dq  = (const float*)d_in[4];
  const float* W_uq  = (const float*)d_in[5];
  const float* W_qr  = (const float*)d_in[6];
  const float* W_uk  = (const float*)d_in[7];
  const float* W_kr  = (const float*)d_in[8];
  const float* W_uv  = (const float*)d_in[9];
  const float* W_o   = (const float*)d_in[10];
  float* out = (float*)d_out;

  char* base = (char*)d_ws;
  size_t off = 0;
  auto nxt = [&](size_t bytes) -> void* {
    void* p = base + off;
    off += (bytes + 255) & ~(size_t)255;
    return p;
  };
  bf16_t* qb    = (bf16_t*)nxt((size_t)B_ * LQ_ * QDIM_ * 2);
  bf16_t* kvb   = (bf16_t*)nxt((size_t)B_ * LK_ * KVDIM_ * 2);
  bf16_t* wkvcT = (bf16_t*)nxt((size_t)KVC_ * KVDIM_ * 2);
  bf16_t* wdqT  = (bf16_t*)nxt((size_t)QC_ * QDIM_ * 2);
  bf16_t* wuqT  = (bf16_t*)nxt((size_t)(H_ * HD_) * QC_ * 2);
  bf16_t* wqrT  = (bf16_t*)nxt((size_t)(H_ * R_) * QC_ * 2);
  bf16_t* wukT  = (bf16_t*)nxt((size_t)(H_ * HD_) * KVC_ * 2);
  bf16_t* wkrT  = (bf16_t*)nxt((size_t)R_ * KVC_ * 2);
  bf16_t* wuvT  = (bf16_t*)nxt((size_t)(H_ * HD_) * KVC_ * 2);
  bf16_t* woT   = (bf16_t*)nxt((size_t)QDIM_ * (H_ * HD_) * 2);
  bf16_t* kvcb  = (bf16_t*)nxt((size_t)B_ * LK_ * KVC_ * 2);
  bf16_t* qcb   = (bf16_t*)nxt((size_t)B_ * LQ_ * QC_ * 2);
  bf16_t* qnb   = (bf16_t*)nxt((size_t)B_ * LQ_ * H_ * HD_ * 2);
  bf16_t* qrb   = (bf16_t*)nxt((size_t)B_ * LQ_ * H_ * R_ * 2);
  bf16_t* knb   = (bf16_t*)nxt((size_t)B_ * LK_ * H_ * HD_ * 2);
  bf16_t* vbb   = (bf16_t*)nxt((size_t)B_ * LK_ * H_ * HD_ * 2);
  bf16_t* krb   = (bf16_t*)nxt((size_t)B_ * LK_ * R_ * 2);
  bf16_t* attnb = (bf16_t*)nxt((size_t)B_ * LQ_ * H_ * HD_ * 2);

  auto cgrid = [](long n) { return dim3((unsigned)((n + 255) / 256)); };

  long nq = (long)B_ * LQ_ * QDIM_, nkv = (long)B_ * LK_ * KVDIM_;
  cast_f32_bf16<<<cgrid(nq), 256, 0, stream>>>(q, qb, nq);
  cast_f32_bf16<<<cgrid(nkv), 256, 0, stream>>>(kv, kvb, nkv);

  auto tc = [&](const float* W, bf16_t* WT, int K, int N) {
    transpose_cast<<<cgrid((long)K * N), 256, 0, stream>>>(W, WT, K, N);
  };
  tc(W_kvc, wkvcT, KVDIM_, KVC_);
  tc(W_dq,  wdqT,  QDIM_,  QC_);
  tc(W_uq,  wuqT,  QC_,    H_ * HD_);
  tc(W_qr,  wqrT,  QC_,    H_ * R_);
  tc(W_uk,  wukT,  KVC_,   H_ * HD_);
  tc(W_kr,  wkrT,  KVC_,   R_);
  tc(W_uv,  wuvT,  KVC_,   H_ * HD_);
  tc(W_o,   woT,   H_ * HD_, QDIM_);

  auto gemm_bf = [&](const bf16_t* A, const bf16_t* Bt, bf16_t* C, int M, int N, int K) {
    dim3 g((N + TILE_N - 1) / TILE_N, (M + TILE_M - 1) / TILE_M);
    gemm_wmma_bf16<true><<<g, 256, 0, stream>>>(A, Bt, (void*)C, M, N, K);
  };
  auto gemm_f32 = [&](const bf16_t* A, const bf16_t* Bt, float* C, int M, int N, int K) {
    dim3 g((N + TILE_N - 1) / TILE_N, (M + TILE_M - 1) / TILE_M);
    gemm_wmma_bf16<false><<<g, 256, 0, stream>>>(A, Bt, (void*)C, M, N, K);
  };

  gemm_bf(kvb,  wkvcT, kvcb, B_ * LK_, KVC_,      KVDIM_);   // kv_c
  gemm_bf(qb,   wdqT,  qcb,  B_ * LQ_, QC_,       QDIM_);    // q_c
  gemm_bf(qcb,  wuqT,  qnb,  B_ * LQ_, H_ * HD_,  QC_);      // q_nope
  gemm_bf(qcb,  wqrT,  qrb,  B_ * LQ_, H_ * R_,   QC_);      // q_rope
  gemm_bf(kvcb, wukT,  knb,  B_ * LK_, H_ * HD_,  KVC_);     // k_nope
  gemm_bf(kvcb, wuvT,  vbb,  B_ * LK_, H_ * HD_,  KVC_);     // v
  gemm_bf(kvcb, wkrT,  krb,  B_ * LK_, R_,        KVC_);     // k_rope

  rope_kernel<<<cgrid((long)B_ * LQ_ * H_ * 32), 256, 0, stream>>>(qrb, B_ * LQ_ * H_, H_, LQ_);
  rope_kernel<<<cgrid((long)B_ * LK_ * 32), 256, 0, stream>>>(krb, B_ * LK_, 1, LK_);

  attn_sparse<<<cgrid((long)B_ * LQ_ * H_ * 32), 256, 0, stream>>>(
      qnb, qrb, knb, krb, vbb, seg, attnb);

  gemm_f32(attnb, woT, out, B_ * LQ_, QDIM_, H_ * HD_);      // out = attn @ W_o
}
